// Hydra_25323127177744
// MI455X (gfx1250) — matrix operand order repaired and checked by reading, not compiled, not run on hardware
//
#include <hip/hip_runtime.h>
#include <hip/hip_bf16.h>

// ---------------------------------------------------------------------------
// Types
// ---------------------------------------------------------------------------
typedef __bf16 bf16;
typedef __attribute__((ext_vector_type(16))) __bf16 v16bf;
typedef __attribute__((ext_vector_type(8)))  __bf16 bf16x8;
typedef __attribute__((ext_vector_type(8)))  float  v8f;
typedef int v4i __attribute__((vector_size(16)));   // matches builtin 'V4i'

#define AS1 __attribute__((address_space(1)))
#define AS3 __attribute__((address_space(3)))

#define D_DIM   1024
#define ROT     512
#define B_DIM   4
#define S_DIM   8192
#define NROWS   (B_DIM * S_DIM)   // 32768
#define K_DIM   1024
#define N1      3072
#define N2      1024

// Async global->LDS path availability (device pass only)
#if defined(__HIP_DEVICE_COMPILE__) && \
    __has_builtin(__builtin_amdgcn_global_load_async_to_lds_b128) && \
    __has_builtin(__builtin_amdgcn_s_wait_asynccnt)
#define USE_ASYNC 1
#else
#define USE_ASYNC 0
#endif

static __device__ inline v16bf frag_combine(bf16x8 lo, bf16x8 hi) {
    return __builtin_shufflevector(lo, hi, 0,1,2,3,4,5,6,7,8,9,10,11,12,13,14,15);
}

static __device__ inline v8f wmma_bf16(v16bf a, v16bf b, v8f c) {
#if defined(__HIP_DEVICE_COMPILE__)
    return __builtin_amdgcn_wmma_f32_16x16x32_bf16(false, a, false, b,
                                                   (short)0, c, false, false);
#else
    return c;
#endif
}

static __device__ inline void async_b128(const bf16* g, bf16* l) {
#if USE_ASYNC
    __builtin_amdgcn_global_load_async_to_lds_b128(
        (AS1 v4i*)(uintptr_t)g,
        (AS3 v4i*)(uint32_t)(uintptr_t)l, 0, 0);
#endif
}

// ---------------------------------------------------------------------------
// fp32 -> bf16 weight conversion (grid-stride)
// ---------------------------------------------------------------------------
__global__ void cvt_bf16_kernel(const float* __restrict__ src, bf16* __restrict__ dst, int n) {
    for (int i = blockIdx.x * blockDim.x + threadIdx.x; i < n; i += gridDim.x * blockDim.x)
        dst[i] = (bf16)src[i];
}

__global__ void zero_kernel(float* __restrict__ p, int n) {
    int i = blockIdx.x * blockDim.x + threadIdx.x;
    if (i < n) p[i] = 0.0f;
}

// ---------------------------------------------------------------------------
// LayerNorm row kernel: one block (256 thr) per row of 1024, writes bf16
// ---------------------------------------------------------------------------
__global__ __launch_bounds__(256) void layernorm_kernel(
    const float* __restrict__ X, const float* __restrict__ g,
    const float* __restrict__ b, bf16* __restrict__ Xn) {
    const int row = blockIdx.x;
    const int t   = threadIdx.x;
    const float* xr = X + (size_t)row * D_DIM;

    __shared__ float s1[256], s2[256];
    float xv[4];
    float lsum = 0.f, lsq = 0.f;
#pragma unroll
    for (int j = 0; j < 4; ++j) {
        float x = xr[t + 256 * j];
        xv[j] = x; lsum += x; lsq += x * x;
    }
    s1[t] = lsum; s2[t] = lsq;
    __syncthreads();
    for (int off = 128; off > 0; off >>= 1) {
        if (t < off) { s1[t] += s1[t + off]; s2[t] += s2[t + off]; }
        __syncthreads();
    }
    const float mu   = s1[0] * (1.0f / D_DIM);
    const float var  = s2[0] * (1.0f / D_DIM) - mu * mu;
    const float rstd = rsqrtf(var + 1e-5f);
    bf16* yr = Xn + (size_t)row * D_DIM;
#pragma unroll
    for (int j = 0; j < 4; ++j) {
        int d = t + 256 * j;
        yr[d] = (bf16)((xv[j] - mu) * rstd * g[d] + b[d]);
    }
}

// ---------------------------------------------------------------------------
// WMMA GEMM:  out[m,n] = sum_k A[m,k] * W[n,k]  (+ residual[m,n])
// A: (M x K) bf16 row-major, W: (N x K) bf16 row-major, out fp32.
// Block tile 128(M) x 128(N); 8 waves (4x2); wave tile 32x64 = 2x4 WMMA frags.
// Async path: double-buffered LDS, global_load_async_to_lds_b128 overlapped
// with WMMA compute, synchronized via s_wait_asynccnt + barrier.
// ---------------------------------------------------------------------------
__global__ __launch_bounds__(256) void wmma_gemm_kernel(
    const bf16* __restrict__ A, const bf16* __restrict__ W,
    const float* __restrict__ residual, float* __restrict__ out,
    int N, int K) {
    const int tid  = threadIdx.x;
    const int wid  = tid >> 5;
    const int lane = tid & 31;
    const int wm   = wid & 3;       // 0..3 -> M (32 rows each)
    const int wn   = wid >> 2;      // 0..1 -> N (64 cols each)
    const int r    = lane & 15;     // row/col within fragment
    const int h    = lane >> 4;     // half-wave selects K sub-range
    const int m0   = blockIdx.x * 128;
    const int n0   = blockIdx.y * 128;

    __shared__ __align__(16) bf16 As[2][128][40];   // 32-wide K slab + pad
    __shared__ __align__(16) bf16 Bs[2][128][40];

    v8f acc[2][4] = {};

    // per-thread staging coords: 128 rows x 4 chunks of 16B, 2 chunks/thread
    const int row0 = tid >> 2;        // 0..63  (second chunk: row0+64)
    const int cc   = (tid & 3) * 8;   // element offset of 16B chunk

    // fragment K sub-ranges (CDNA5 16-bit WMMA VGPR layouts)
    const int klo = 8 * h;            // A: h=0 -> K{0..7,16..23}; h=1 -> {8..15,24..31}
    const int khi = 16 + 8 * h;
    const int kb  = 8 * h;            // B: h=0 -> K{0..7,16..23}; h=1 -> {8..15,24..31}

    auto compute = [&](int buf) {
        v16bf fa[2], fb[4];
#pragma unroll
        for (int tm = 0; tm < 2; ++tm) {
            int row = wm * 32 + tm * 16 + r;
            fa[tm] = frag_combine(*(const bf16x8*)&As[buf][row][klo],
                                  *(const bf16x8*)&As[buf][row][khi]);
        }
#pragma unroll
        for (int tn = 0; tn < 4; ++tn) {
            int col = wn * 64 + tn * 16 + r;
            fb[tn] = frag_combine(*(const bf16x8*)&Bs[buf][col][kb],
                                  *(const bf16x8*)&Bs[buf][col][kb + 16]);
        }
#pragma unroll
        for (int tm = 0; tm < 2; ++tm)
#pragma unroll
            for (int tn = 0; tn < 4; ++tn)
                acc[tm][tn] = wmma_bf16(fa[tm], fb[tn], acc[tm][tn]);
    };

#if USE_ASYNC
    auto issue = [&](int k0, int buf) {
        const bf16* ga = A + (size_t)(m0 + row0) * K + k0 + cc;
        const bf16* gb = W + (size_t)(n0 + row0) * K + k0 + cc;
        async_b128(ga,                  &As[buf][row0][cc]);
        async_b128(ga + (size_t)64 * K, &As[buf][row0 + 64][cc]);
        async_b128(gb,                  &Bs[buf][row0][cc]);
        async_b128(gb + (size_t)64 * K, &Bs[buf][row0 + 64][cc]);
    };

    const int nslab = K >> 5;
    issue(0, 0);
    for (int ks = 0; ks < nslab; ++ks) {
        const int buf = ks & 1;
        __builtin_amdgcn_s_wait_asynccnt(0);   // slab ks landed in LDS
        __syncthreads();                       // visible to all waves
        if (ks + 1 < nslab)
            issue((ks + 1) << 5, buf ^ 1);     // overlap with compute below
        compute(buf);
        __syncthreads();                       // all reads of buf done
    }
#else
    // Fallback: synchronous register-staged single-buffer pipeline
    for (int k0 = 0; k0 < K; k0 += 32) {
        bf16x8 ra0 = *(const bf16x8*)(A + (size_t)(m0 + row0) * K + k0 + cc);
        bf16x8 ra1 = *(const bf16x8*)(A + (size_t)(m0 + row0 + 64) * K + k0 + cc);
        bf16x8 rb0 = *(const bf16x8*)(W + (size_t)(n0 + row0) * K + k0 + cc);
        bf16x8 rb1 = *(const bf16x8*)(W + (size_t)(n0 + row0 + 64) * K + k0 + cc);
        __syncthreads();
        *(bf16x8*)&As[0][row0][cc]      = ra0;
        *(bf16x8*)&As[0][row0 + 64][cc] = ra1;
        *(bf16x8*)&Bs[0][row0][cc]      = rb0;
        *(bf16x8*)&Bs[0][row0 + 64][cc] = rb1;
        __syncthreads();
        if (k0 + 32 < K) {
            __builtin_prefetch(A + (size_t)(m0 + row0) * K + k0 + 32, 0, 1);
            __builtin_prefetch(W + (size_t)(n0 + row0) * K + k0 + 32, 0, 1);
        }
        compute(0);
    }
#endif

    // ---- epilogue: C/D layout — lane = N col (r), VGPR p -> M = p + 8*h ----
#pragma unroll
    for (int tm = 0; tm < 2; ++tm)
#pragma unroll
        for (int tn = 0; tn < 4; ++tn) {
            int n = n0 + wn * 64 + tn * 16 + r;
#pragma unroll
            for (int p = 0; p < 8; ++p) {
                int m = m0 + wm * 32 + tm * 16 + p + 8 * h;
                float v = acc[tm][tn][p];
                size_t idx = (size_t)m * N + n;
                if (residual) v += residual[idx];
                out[idx] = v;
            }
        }
}

// ---------------------------------------------------------------------------
// Rotary(Q in place) + Rotary(K) + A[b,:] += l2norm(K) * V  over 128 rows.
// grid: (B, S/128); block: 256
// ---------------------------------------------------------------------------
__global__ __launch_bounds__(256) void rotary_reduce_kernel(
    float* __restrict__ QKV, float* __restrict__ Avec) {
    const int b     = blockIdx.x;
    const int chunk = blockIdx.y;
    const int t     = threadIdx.x;

    __shared__ float Krow[D_DIM];
    __shared__ float red[256];

    // inv_freq_t = 10000^(-t/256)  -> theta = s * exp(t * c)
    const float cexp = -0.0359778919f;   // -ln(10000)/256
    const float invf = __expf(cexp * (float)t);

    float acc[4] = {0.f, 0.f, 0.f, 0.f};

    for (int rr = 0; rr < 128; ++rr) {
        const int s = chunk * 128 + rr;
        float* base = QKV + (size_t)(b * S_DIM + s) * N1;
        float* Q = base;
        float* Kp = base + D_DIM;
        const float* V = base + 2 * D_DIM;

        float sn, cs;
        __sincosf((float)s * invf, &sn, &cs);

        // rotary on Q, in place (pair t of 256)
        {
            float q0 = Q[2 * t], q1 = Q[2 * t + 1];
            Q[2 * t]     = q0 * cs - q1 * sn;
            Q[2 * t + 1] = q1 * cs + q0 * sn;
        }
        // rotary on K into LDS; top half copied unrotated
        {
            float k0 = Kp[2 * t], k1 = Kp[2 * t + 1];
            Krow[2 * t]     = k0 * cs - k1 * sn;
            Krow[2 * t + 1] = k1 * cs + k0 * sn;
            Krow[ROT + t]        = Kp[ROT + t];
            Krow[ROT + 256 + t]  = Kp[ROT + 256 + t];
        }
        __syncthreads();

        // sum of squares of K row
        float lsq = 0.f;
#pragma unroll
        for (int j = 0; j < 4; ++j) {
            float k = Krow[t + 256 * j];
            lsq += k * k;
        }
        red[t] = lsq;
        __syncthreads();
        for (int off = 128; off > 0; off >>= 1) {
            if (t < off) red[t] += red[t + off];
            __syncthreads();
        }
        const float rn = rsqrtf(red[0] + 1e-6f);

        // accumulate l2norm(K) * V
#pragma unroll
        for (int j = 0; j < 4; ++j) {
            int d = t + 256 * j;
            acc[j] += Krow[d] * rn * V[d];
        }
        __syncthreads();   // Krow free for next row
    }

#pragma unroll
    for (int j = 0; j < 4; ++j)
        atomicAdd(&Avec[b * D_DIM + t + 256 * j], acc[j]);
}

// ---------------------------------------------------------------------------
// Y[row,:] = A[b,:] * l2norm(Q_rot[row,:])  -> bf16 ; one block per row
// ---------------------------------------------------------------------------
__global__ __launch_bounds__(256) void scale_q_kernel(
    const float* __restrict__ QKV, const float* __restrict__ Avec,
    bf16* __restrict__ Yb) {
    const int row = blockIdx.x;
    const int b   = row >> 13;           // /8192
    const int t   = threadIdx.x;
    const float* Q = QKV + (size_t)row * N1;

    __shared__ float red[256];
    float qv[4];
    float lsq = 0.f;
#pragma unroll
    for (int j = 0; j < 4; ++j) {
        float q = Q[t + 256 * j];
        qv[j] = q; lsq += q * q;
    }
    red[t] = lsq;
    __syncthreads();
    for (int off = 128; off > 0; off >>= 1) {
        if (t < off) red[t] += red[t + off];
        __syncthreads();
    }
    const float rn = rsqrtf(red[0] + 1e-6f);
    bf16* yr = Yb + (size_t)row * D_DIM;
#pragma unroll
    for (int j = 0; j < 4; ++j) {
        int d = t + 256 * j;
        yr[d] = (bf16)(Avec[b * D_DIM + d] * qv[j] * rn);
    }
}

// ---------------------------------------------------------------------------
// Launch
// ---------------------------------------------------------------------------
extern "C" void kernel_launch(void* const* d_in, const int* in_sizes, int n_in,
                              void* d_out, int out_size, void* d_ws, size_t ws_size,
                              hipStream_t stream) {
    const float* X     = (const float*)d_in[0];
    const float* W_in  = (const float*)d_in[1];
    const float* W_out = (const float*)d_in[2];
    const float* ln_g  = (const float*)d_in[3];
    const float* ln_b  = (const float*)d_in[4];
    float* out = (float*)d_out;

    char* ws = (char*)d_ws;
    // workspace layout (bytes)
    bf16*  Xnb   = (bf16*)(ws);                       //  64 MiB
    bf16*  Winb  = (bf16*)(ws + 67108864);            //   6 MiB
    bf16*  Woutb = (bf16*)(ws + 73400320);            //   2 MiB
    float* Avec  = (float*)(ws + 75497472);           //  16 KiB
    bf16*  Yb    = (bf16*)(ws + 75513856);            //  64 MiB
    float* QKV   = (float*)(ws + 142622720);          // 384 MiB

    // 1) weights -> bf16
    cvt_bf16_kernel<<<4096, 256, 0, stream>>>(W_in,  Winb,  N1 * K_DIM);
    cvt_bf16_kernel<<<2048, 256, 0, stream>>>(W_out, Woutb, N2 * K_DIM);
    // 2) LayerNorm -> bf16 Xn
    layernorm_kernel<<<NROWS, 256, 0, stream>>>(X, ln_g, ln_b, Xnb);
    // 3) zero A accumulator
    zero_kernel<<<16, 256, 0, stream>>>(Avec, B_DIM * D_DIM);
    // 4) GEMM1: QKV = Xn @ W_in^T   (M=32768, N=3072, K=1024)
    wmma_gemm_kernel<<<dim3(NROWS / 128, N1 / 128), 256, 0, stream>>>(
        Xnb, Winb, nullptr, QKV, N1, K_DIM);
    // 5) rotary(Q,K) + A = sum_s l2norm(K)*V
    rotary_reduce_kernel<<<dim3(B_DIM, S_DIM / 128), 256, 0, stream>>>(QKV, Avec);
    // 6) Y = A * l2norm(Q) -> bf16
    scale_q_kernel<<<NROWS, 256, 0, stream>>>(QKV, Avec, Yb);
    // 7) GEMM2: out = Y @ W_out^T + X   (M=32768, N=1024, K=1024)
    wmma_gemm_kernel<<<dim3(NROWS / 128, N2 / 128), 256, 0, stream>>>(
        Yb, Woutb, X, out, N2, K_DIM);
}
